// EncoderLayer_87067577024713
// MI455X (gfx1250) — compile-verified, matrix operand order checked by reading, not run on hardware
//
#include <hip/hip_runtime.h>

#define B_  8
#define S_  1024
#define D_  2048
#define H_  16
#define FF_ 8192
#define DH_ 128

typedef __bf16 bf16_t;
typedef bf16_t v16bf __attribute__((ext_vector_type(16)));
typedef float  v8f   __attribute__((ext_vector_type(8)));
typedef unsigned int uv4 __attribute__((ext_vector_type(4)));

struct U8x { uv4 a, b; };
union HV { uv4 v; unsigned short h[8]; };

__device__ __forceinline__ unsigned short f32_to_bf16(float f) {
  unsigned u = __float_as_uint(f);
  u += 0x7FFFu + ((u >> 16) & 1u);           // round to nearest even
  return (unsigned short)(u >> 16);
}

__device__ __forceinline__ unsigned pack_bf16x2(float a, float b) {
  return (unsigned)f32_to_bf16(a) | ((unsigned)f32_to_bf16(b) << 16);
}

// Per-lane async DMA: 16 bytes global -> LDS, tracked by ASYNCcnt.
// LDS address operand is the low 32 bits of the generic pointer
// (ISA 10.2).
__device__ __forceinline__ void async_copy_b128(const void* gsrc, void* ldst) {
  const unsigned loff = (unsigned)(size_t)ldst;
  asm volatile("global_load_async_to_lds_b128 %0, %1, off"
               :: "v"(loff), "v"(gsrc)
               : "memory");
}
__device__ __forceinline__ void wait_async() {
  asm volatile("s_wait_asynccnt 0x0" ::: "memory");
}

// Load a 16x32 bf16 WMMA fragment (A or B operand) from an LDS tile stored
// row-major with row length `ld` (bf16 elements). Per ISA 7.12.2:
//   lane<16 : row = lane,    K = {0..7, 16..23}
//   lane>=16: row = lane-16, K = {8..15, 24..31}
__device__ __forceinline__ v16bf frag_lds(const unsigned short* t, int ld, int lane, int kbase) {
  const int r  = lane & 15;
  const int ko = kbase + ((lane >> 4) << 3);
  const unsigned short* p = t + r * ld + ko;
  U8x u;
  u.a = *(const uv4*)(p);
  u.b = *(const uv4*)(p + 16);
  return __builtin_bit_cast(v16bf, u);
}

// ---------------------------------------------------------------------------
// fp32 -> bf16 conversion (float4 in, packed uint2 out). n4 = n/4.
// ---------------------------------------------------------------------------
__global__ __launch_bounds__(256) void f2bf_kernel(const float* __restrict__ in,
                                                   unsigned short* __restrict__ out,
                                                   size_t n4) {
  size_t i = (size_t)blockIdx.x * 256 + threadIdx.x;
  const size_t stride = (size_t)gridDim.x * 256;
  const float4* in4 = (const float4*)in;
  for (; i < n4; i += stride) {
    const float4 v = in4[i];
    uint2 pk;
    pk.x = pack_bf16x2(v.x, v.y);
    pk.y = pack_bf16x2(v.z, v.w);
    *(uint2*)&out[i * 4] = pk;
  }
}

// ---------------------------------------------------------------------------
// Tiled bf16 GEMM:  C[M,N] = A[M,K] * Bw[N,K]^T + bias [+ resid] [ReLU]
// Block = 256 threads (8 waves), tile 128x128, K-step 32.
// Async global->LDS DMA double-buffering: WMMA on buffer `cur` overlaps the
// DMA into buffer `cur^1`; drain ASYNCcnt just before the barrier.
// ---------------------------------------------------------------------------
#define BM 128
#define BN 128
#define BK 32
#define LDT 40   // 32 + 8 bf16 pad (16B) -> conflict-free ds_load_b128

// reuse_a is set on all but the last WMMA of each shared-A run (constant int
// required by the builtin).
#define GEMM_WMMA(i, j, ra)                                                   \
  acc[i][j] = __builtin_amdgcn_wmma_f32_16x16x32_bf16(                        \
      false, af[i], false, bfr[j], (short)0, acc[i][j], ra, false)

__global__ __launch_bounds__(256) void gemm_bf16_kernel(
    const unsigned short* __restrict__ A, const unsigned short* __restrict__ Bw,
    const float* __restrict__ bias, const float* __restrict__ resid,
    unsigned short* __restrict__ outb, float* __restrict__ outf,
    int M, int N, int K, int relu)
{
  __shared__ unsigned short As[2][BM * LDT];
  __shared__ unsigned short Bs[2][BN * LDT];
  const int tid  = threadIdx.x;
  const int lane = tid & 31;
  const int wave = tid >> 5;
  const int m0 = blockIdx.y * BM;
  const int n0 = blockIdx.x * BN;
  const int wm = (wave & 3) * 32;
  const int wn = (wave >> 2) * 64;

  // Staging: 512 chunks of 8 bf16 per tile; each thread owns 2 fixed chunks.
  const int row0 = tid >> 2;          // 0..63
  const int col0 = (tid & 3) << 3;    // 0,8,16,24
  const unsigned short* aP0 = A  + (size_t)(m0 + row0) * K + col0;
  const unsigned short* aP1 = A  + (size_t)(m0 + row0 + 64) * K + col0;
  const unsigned short* bP0 = Bw + (size_t)(n0 + row0) * K + col0;
  const unsigned short* bP1 = Bw + (size_t)(n0 + row0 + 64) * K + col0;
  const int s0 = row0 * LDT + col0;
  const int s1 = (row0 + 64) * LDT + col0;

  const v8f vzero = {0.f, 0.f, 0.f, 0.f, 0.f, 0.f, 0.f, 0.f};
  v8f acc[2][4];
#pragma unroll
  for (int i = 0; i < 2; ++i)
#pragma unroll
    for (int j = 0; j < 4; ++j) acc[i][j] = vzero;

  // Prologue: DMA tile 0 into buffer 0.
  async_copy_b128(aP0, &As[0][s0]);
  async_copy_b128(aP1, &As[0][s1]);
  async_copy_b128(bP0, &Bs[0][s0]);
  async_copy_b128(bP1, &Bs[0][s1]);
  aP0 += BK; aP1 += BK; bP0 += BK; bP1 += BK;
  wait_async();
  __syncthreads();

  int cur = 0;
  for (int k0 = BK; k0 <= K; k0 += BK) {
    const bool more = (k0 < K);
    if (more) {
      const int nxt = cur ^ 1;
      async_copy_b128(aP0, &As[nxt][s0]);
      async_copy_b128(aP1, &As[nxt][s1]);
      async_copy_b128(bP0, &Bs[nxt][s0]);
      async_copy_b128(bP1, &Bs[nxt][s1]);
      aP0 += BK; aP1 += BK; bP0 += BK; bP1 += BK;
    }

    // Compute on current LDS buffer while the DMA fills the other one.
    const unsigned short* Ac = As[cur];
    const unsigned short* Bc = Bs[cur];
    v16bf af[2], bfr[4];
#pragma unroll
    for (int i = 0; i < 2; ++i) af[i] = frag_lds(Ac + (wm + i * 16) * LDT, LDT, lane, 0);
#pragma unroll
    for (int j = 0; j < 4; ++j) bfr[j] = frag_lds(Bc + (wn + j * 16) * LDT, LDT, lane, 0);
    GEMM_WMMA(0, 0, true);
    GEMM_WMMA(0, 1, true);
    GEMM_WMMA(0, 2, true);
    GEMM_WMMA(0, 3, false);
    GEMM_WMMA(1, 0, true);
    GEMM_WMMA(1, 1, true);
    GEMM_WMMA(1, 2, true);
    GEMM_WMMA(1, 3, false);

    if (more) wait_async();
    __syncthreads();
    cur ^= 1;
  }

  // Epilogue. C layout: lane<16 -> M=r, N=lane; lane>=16 -> M=r+8, N=lane-16.
  const int r_ = lane & 15;
  const int hi = lane >> 4;
#pragma unroll
  for (int i = 0; i < 2; ++i) {
#pragma unroll
    for (int j = 0; j < 4; ++j) {
      const int col = n0 + wn + j * 16 + r_;
      const float bv = bias ? bias[col] : 0.0f;
#pragma unroll
      for (int r = 0; r < 8; ++r) {
        const int row = m0 + wm + i * 16 + r + hi * 8;
        const size_t idx = (size_t)row * N + col;
        float v = acc[i][j][r] + bv;
        if (resid) v += resid[idx];
        if (relu) v = fmaxf(v, 0.0f);
        if (outf) outf[idx] = v;
        if (outb) outb[idx] = f32_to_bf16(v);
      }
    }
  }
}

// ---------------------------------------------------------------------------
// Flash attention (bf16 WMMA, fp32 online softmax).
// grid = (S/128, B*H), block = 256 (8 waves). Each wave owns a 16-row Q tile.
// K tile staged via async DMA; V staged transposed through registers.
// ---------------------------------------------------------------------------
#define PV_WMMA(t, ra)                                                        \
  {                                                                           \
    v16bf vf = frag_lds(&Vt[(t * 16) * 40], 40, lane, 0);                     \
    o[t] = __builtin_amdgcn_wmma_f32_16x16x32_bf16(                           \
        false, pf, false, vf, (short)0, o[t], ra, false);                     \
  }

__global__ __launch_bounds__(256) void attn_kernel(
    const unsigned short* __restrict__ Qb, const unsigned short* __restrict__ Kb,
    const unsigned short* __restrict__ Vb, const int* __restrict__ mask,
    unsigned short* __restrict__ Ob)
{
  __shared__ unsigned short Kl[32 * 136];   // [key][dh], pad 128->136
  __shared__ unsigned short Vt[128 * 40];   // [dh][key], pad 32->40
  __shared__ unsigned short Pl[8][16 * 40]; // per-wave P staging, 16x32 pad->40
  const int tid  = threadIdx.x;
  const int lane = tid & 31;
  const int wave = tid >> 5;
  const int bh = blockIdx.y;
  const int b = bh / H_;
  const int h = bh % H_;
  const int q0 = blockIdx.x * 128 + wave * 16;
  const size_t headOff = (size_t)b * S_ * D_ + (size_t)h * DH_;
  const unsigned short* Qh = Qb + headOff;
  const unsigned short* Kh = Kb + headOff;
  const unsigned short* Vh = Vb + headOff;
  unsigned short* Oh = Ob + headOff;

  const int r_ = lane & 15;
  const int hi = lane >> 4;

  // Q fragments (4 K-chunks of 32 over Dh=128), loaded straight from global.
  v16bf qf[4];
#pragma unroll
  for (int c = 0; c < 4; ++c) {
    const unsigned short* p = Qh + (size_t)(q0 + r_) * D_ + c * 32 + hi * 8;
    U8x u;
    u.a = *(const uv4*)(p);
    u.b = *(const uv4*)(p + 16);
    qf[c] = __builtin_bit_cast(v16bf, u);
  }

  // Per-row mask pointers (rows fixed per lane for the whole kernel).
  const int* mrp[8];
#pragma unroll
  for (int r = 0; r < 8; ++r) mrp[r] = mask + (size_t)(q0 + r + hi * 8) * S_;

  const v8f vzero = {0.f, 0.f, 0.f, 0.f, 0.f, 0.f, 0.f, 0.f};
  float mrow[8], srow[8];
  v8f o[8];
#pragma unroll
  for (int r = 0; r < 8; ++r) { mrow[r] = -1e30f; srow[r] = 0.f; }
#pragma unroll
  for (int t = 0; t < 8; ++t) o[t] = vzero;

  const float scale = 0.088388347648318447f;  // 1/sqrt(Dh)

  for (int kt = 0; kt < S_; kt += 32) {
    // Stage K tile (32x128 bf16) via async DMA: 512 x 16B chunks, 2/thread.
#pragma unroll
    for (int i = 0; i < 2; ++i) {
      const int c = tid + (i << 8);
      const int row = c >> 4;
      const int col = (c & 15) << 3;
      async_copy_b128(&Kh[(size_t)(kt + row) * D_ + col], &Kl[row * 136 + col]);
    }
    // Stage V tile transposed: vector read of 8 dh values, scalar LDS scatter.
#pragma unroll
    for (int i = 0; i < 2; ++i) {
      const int c = tid + (i << 8);
      const int key = c >> 4;         // 0..31
      const int dhc = (c & 15) << 3;  // 0,8,...,120
      HV u;
      u.v = *(const uv4*)&Vh[(size_t)(kt + key) * D_ + dhc];
#pragma unroll
      for (int e = 0; e < 8; ++e) Vt[(dhc + e) * 40 + key] = u.h[e];
    }
    if (kt + 32 < S_) {
      __builtin_prefetch(&Vh[(size_t)(kt + 32 + (tid >> 3)) * D_], 0, 1);
    }
    wait_async();
    __syncthreads();

    // Scores: S = Q(16x128) * K^T(128x32), two 16-key column tiles.
    v8f scv[2];
#pragma unroll
    for (int j = 0; j < 2; ++j) {
      v8f z = vzero;
#pragma unroll
      for (int c = 0; c < 4; ++c) {
        v16bf kf = frag_lds(&Kl[(j * 16) * 136], 136, lane, c * 32);
        z = __builtin_amdgcn_wmma_f32_16x16x32_bf16(
            false, qf[c], false, kf, (short)0, z, false, false);
      }
      scv[j] = z;
    }

    // Online softmax per row (rows split across half-waves; width-16 shuffles).
    float pscale[8];
#pragma unroll
    for (int r = 0; r < 8; ++r) {
      float s0 = scv[0][r] * scale;
      float s1 = scv[1][r] * scale;
      if (mrp[r][kt + r_] == 0)      s0 = -1e9f;
      if (mrp[r][kt + 16 + r_] == 0) s1 = -1e9f;
      float mx = fmaxf(s0, s1);
#pragma unroll
      for (int off = 8; off > 0; off >>= 1) mx = fmaxf(mx, __shfl_xor(mx, off, 16));
      const float mnew = fmaxf(mrow[r], mx);
      const float corr = __expf(mrow[r] - mnew);
      const float p0 = __expf(s0 - mnew);
      const float p1 = __expf(s1 - mnew);
      float psum = p0 + p1;
#pragma unroll
      for (int off = 8; off > 0; off >>= 1) psum += __shfl_xor(psum, off, 16);
      srow[r] = srow[r] * corr + psum;
      mrow[r] = mnew;
      pscale[r] = corr;
      unsigned short* pp = &Pl[wave][(r + hi * 8) * 40];
      pp[r_]      = f32_to_bf16(p0);
      pp[r_ + 16] = f32_to_bf16(p1);
    }

#pragma unroll
    for (int t = 0; t < 8; ++t)
#pragma unroll
      for (int r = 0; r < 8; ++r) o[t][r] *= pscale[r];

    // O += P(16x32) * V(32x128): 8 column tiles, shared A operand -> reuse_a.
    const v16bf pf = frag_lds(&Pl[wave][0], 40, lane, 0);
    PV_WMMA(0, true);
    PV_WMMA(1, true);
    PV_WMMA(2, true);
    PV_WMMA(3, true);
    PV_WMMA(4, true);
    PV_WMMA(5, true);
    PV_WMMA(6, true);
    PV_WMMA(7, false);
    __syncthreads();
  }

#pragma unroll
  for (int r = 0; r < 8; ++r) srow[r] = 1.0f / srow[r];
#pragma unroll
  for (int t = 0; t < 8; ++t) {
#pragma unroll
    for (int r = 0; r < 8; ++r) {
      const int row = q0 + r + hi * 8;
      const int col = t * 16 + r_;
      Oh[(size_t)row * D_ + col] = f32_to_bf16(o[t][r] * srow[r]);
    }
  }
}

// ---------------------------------------------------------------------------
// Row LayerNorm over D_ (=2048). One block (256 threads) per row, float4 I/O.
// ---------------------------------------------------------------------------
__global__ __launch_bounds__(256) void layernorm_kernel(
    const float* __restrict__ in, const float* __restrict__ gamma,
    const float* __restrict__ beta, float* __restrict__ outf,
    unsigned short* __restrict__ outb)
{
  const int row = blockIdx.x;
  const float* x = in + (size_t)row * D_;
  const float4* x4 = (const float4*)x;
  float4 vv[2];
  float s = 0.f, ss = 0.f;
#pragma unroll
  for (int j = 0; j < 2; ++j) {
    vv[j] = x4[threadIdx.x + j * 256];
    s  += vv[j].x + vv[j].y + vv[j].z + vv[j].w;
    ss += vv[j].x * vv[j].x + vv[j].y * vv[j].y + vv[j].z * vv[j].z + vv[j].w * vv[j].w;
  }
#pragma unroll
  for (int off = 16; off > 0; off >>= 1) {
    s  += __shfl_xor(s, off, 32);
    ss += __shfl_xor(ss, off, 32);
  }
  __shared__ float ps[8], pss[8];
  __shared__ float meanS, rstdS;
  if ((threadIdx.x & 31) == 0) { ps[threadIdx.x >> 5] = s; pss[threadIdx.x >> 5] = ss; }
  __syncthreads();
  if (threadIdx.x == 0) {
    float t1 = 0.f, t2 = 0.f;
#pragma unroll
    for (int i = 0; i < 8; ++i) { t1 += ps[i]; t2 += pss[i]; }
    const float mean = t1 * (1.0f / D_);
    const float var  = t2 * (1.0f / D_) - mean * mean;
    meanS = mean;
    rstdS = rsqrtf(var + 1e-5f);
  }
  __syncthreads();
  const float mean = meanS, rstd = rstdS;
#pragma unroll
  for (int j = 0; j < 2; ++j) {
    const int i4 = threadIdx.x + j * 256;
    const float4 g4 = ((const float4*)gamma)[i4];
    const float4 b4 = ((const float4*)beta)[i4];
    float4 o4;
    o4.x = (vv[j].x - mean) * rstd * g4.x + b4.x;
    o4.y = (vv[j].y - mean) * rstd * g4.y + b4.y;
    o4.z = (vv[j].z - mean) * rstd * g4.z + b4.z;
    o4.w = (vv[j].w - mean) * rstd * g4.w + b4.w;
    if (outf) ((float4*)(outf + (size_t)row * D_))[i4] = o4;
    if (outb) {
      uint2 pk;
      pk.x = pack_bf16x2(o4.x, o4.y);
      pk.y = pack_bf16x2(o4.z, o4.w);
      *(uint2*)&outb[(size_t)row * D_ + (size_t)i4 * 4] = pk;
    }
  }
}

// ---------------------------------------------------------------------------
// Host-side orchestration
// ---------------------------------------------------------------------------
extern "C" void kernel_launch(void* const* d_in, const int* in_sizes, int n_in,
                              void* d_out, int out_size, void* d_ws, size_t ws_size,
                              hipStream_t stream)
{
  (void)in_sizes; (void)n_in; (void)out_size; (void)ws_size;
  const float* x    = (const float*)d_in[0];
  const int*   mask = (const int*)d_in[1];
  const float* Wq   = (const float*)d_in[2];
  const float* bq   = (const float*)d_in[3];
  const float* Wk   = (const float*)d_in[4];
  const float* bk   = (const float*)d_in[5];
  const float* Wv   = (const float*)d_in[6];
  const float* bv   = (const float*)d_in[7];
  const float* Wo   = (const float*)d_in[8];
  const float* bo   = (const float*)d_in[9];
  const float* W1   = (const float*)d_in[10];
  const float* b1   = (const float*)d_in[11];
  const float* W2   = (const float*)d_in[12];
  const float* b2   = (const float*)d_in[13];
  const float* g1   = (const float*)d_in[14];
  const float* be1  = (const float*)d_in[15];
  const float* g2   = (const float*)d_in[16];
  const float* be2  = (const float*)d_in[17];
  float* out = (float*)d_out;

  const size_t nXD = (size_t)B_ * S_ * D_;
  const size_t nDD = (size_t)D_ * D_;
  const size_t nDF = (size_t)D_ * FF_;
  const size_t nXF = (size_t)B_ * S_ * FF_;

  char* p = (char*)d_ws;
  auto alloc = [&](size_t bytes) -> void* {
    void* q = (void*)p;
    p += (bytes + 255) & ~(size_t)255;
    return q;
  };
  unsigned short* Xb   = (unsigned short*)alloc(nXD * 2);
  unsigned short* Wqb  = (unsigned short*)alloc(nDD * 2);
  unsigned short* Wkb  = (unsigned short*)alloc(nDD * 2);
  unsigned short* Wvb  = (unsigned short*)alloc(nDD * 2);
  unsigned short* Wob  = (unsigned short*)alloc(nDD * 2);
  unsigned short* W1b  = (unsigned short*)alloc(nDF * 2);
  unsigned short* W2b  = (unsigned short*)alloc(nDF * 2);
  unsigned short* Qbuf = (unsigned short*)alloc(nXD * 2);
  unsigned short* Kbuf = (unsigned short*)alloc(nXD * 2);
  unsigned short* Vbuf = (unsigned short*)alloc(nXD * 2);
  unsigned short* Obuf = (unsigned short*)alloc(nXD * 2);
  float*          x1p  = (float*)alloc(nXD * 4);
  float*          x1f  = (float*)alloc(nXD * 4);
  unsigned short* x1b  = (unsigned short*)alloc(nXD * 2);
  unsigned short* hb   = (unsigned short*)alloc(nXF * 2);
  float*          ffnp = (float*)alloc(nXD * 4);

  const dim3 blk(256);

  // --- fp32 -> bf16 conversion of activations and weights ---
  f2bf_kernel<<<2048, blk, 0, stream>>>(x,  Xb,  nXD / 4);
  f2bf_kernel<<<1024, blk, 0, stream>>>(Wq, Wqb, nDD / 4);
  f2bf_kernel<<<1024, blk, 0, stream>>>(Wk, Wkb, nDD / 4);
  f2bf_kernel<<<1024, blk, 0, stream>>>(Wv, Wvb, nDD / 4);
  f2bf_kernel<<<1024, blk, 0, stream>>>(Wo, Wob, nDD / 4);
  f2bf_kernel<<<2048, blk, 0, stream>>>(W1, W1b, nDF / 4);
  f2bf_kernel<<<2048, blk, 0, stream>>>(W2, W2b, nDF / 4);

  const int M = B_ * S_;
  const dim3 gDD(D_ / BN, M / BM);   // (16, 64)
  const dim3 gDF(FF_ / BN, M / BM);  // (64, 64)

  // --- QKV projections ---
  gemm_bf16_kernel<<<gDD, blk, 0, stream>>>(Xb, Wqb, bq, nullptr, Qbuf, nullptr, M, D_, D_, 0);
  gemm_bf16_kernel<<<gDD, blk, 0, stream>>>(Xb, Wkb, bk, nullptr, Kbuf, nullptr, M, D_, D_, 0);
  gemm_bf16_kernel<<<gDD, blk, 0, stream>>>(Xb, Wvb, bv, nullptr, Vbuf, nullptr, M, D_, D_, 0);

  // --- attention ---
  attn_kernel<<<dim3(S_ / 128, B_ * H_), blk, 0, stream>>>(Qbuf, Kbuf, Vbuf, mask, Obuf);

  // --- output projection + residual, then LN1 ---
  gemm_bf16_kernel<<<gDD, blk, 0, stream>>>(Obuf, Wob, bo, x, nullptr, x1p, M, D_, D_, 0);
  layernorm_kernel<<<M, blk, 0, stream>>>(x1p, g1, be1, x1f, x1b);

  // --- FFN ---
  gemm_bf16_kernel<<<gDF, blk, 0, stream>>>(x1b, W1b, b1, nullptr, hb, nullptr, M, FF_, D_, 1);
  gemm_bf16_kernel<<<gDD, blk, 0, stream>>>(hb, W2b, b2, x1f, nullptr, ffnp, M, D_, FF_, 0);

  // --- LN2 -> final output ---
  layernorm_kernel<<<M, blk, 0, stream>>>(ffnp, g2, be2, out, nullptr);
}
